// MultiHeadSelfAttention_v1_4312147165476
// MI455X (gfx1250) — matrix operand order repaired and checked by reading, not compiled, not run on hardware
//
#include <hip/hip_runtime.h>
#include <hip/hip_bf16.h>

// ---------------------------------------------------------------------------
// Fused MHSA forward for MI455X (gfx1250), wave32 + WMMA f16->f32.
// LN(f32->f16) -> QKV gemm (WMMA, async-LDS double buffer) -> flash attention
// (WMMA, async-LDS K tiles) -> projection gemm -> f32 out.
// ---------------------------------------------------------------------------

typedef __attribute__((ext_vector_type(16))) _Float16 v16h;
typedef __attribute__((ext_vector_type(8)))  float    v8f;

constexpr int Bz  = 2;
constexpr int Nn  = 2048;
constexpr int Cc  = 1024;
constexpr int Hh  = 16;
constexpr int DHh = 64;
constexpr int BNr = Bz * Nn;          // 4096 rows
constexpr int C3  = 3 * Cc;           // 3072

union FragH {           // one 16x(K=32) f16 fragment = 16 halves per lane
    v16h v;
    uint4 q[2];
};

static __device__ __forceinline__ v8f wmma_f16(v16h a, v16h b, v8f c) {
    return __builtin_amdgcn_wmma_f32_16x16x32_f16(false, a, false, b,
                                                  (short)0, c, false, false);
}

// generic shared pointer -> LDS byte offset (ISA: LDS addr = addr[31:0])
static __device__ __forceinline__ unsigned lds_off(const void* p) {
    return (unsigned)(unsigned long long)p;
}
// async global -> LDS 128-bit copy (CDNA5, ASYNCcnt-tracked)
static __device__ __forceinline__ void async_ld_b128(const void* gsrc,
                                                     unsigned ldsdst) {
    asm volatile("global_load_async_to_lds_b128 %0, %1, off"
                 :: "v"(ldsdst), "v"(gsrc) : "memory");
}
static __device__ __forceinline__ void wait_async0() {
    asm volatile("s_wait_asynccnt 0x0" ::: "memory");
}

// ------------------------------- LayerNorm ---------------------------------
__device__ __forceinline__ float block_sum256(float v, float* sbuf) {
    int tid = threadIdx.x;
    for (int m = 16; m; m >>= 1) v += __shfl_xor(v, m, 32);
    if ((tid & 31) == 0) sbuf[tid >> 5] = v;
    __syncthreads();
    if (tid < 32) {
        float t = (tid < 8) ? sbuf[tid] : 0.0f;
        for (int m = 4; m; m >>= 1) t += __shfl_xor(t, m, 32);
        if (tid == 0) sbuf[0] = t;
    }
    __syncthreads();
    float r = sbuf[0];
    __syncthreads();
    return r;
}

__global__ __launch_bounds__(256) void k_layernorm(
    const float* __restrict__ x, const float* __restrict__ gamma,
    const float* __restrict__ beta, _Float16* __restrict__ xn)
{
    __shared__ float sbuf[8];
    const int row = blockIdx.x;                 // [0, 4096)
    const float* xr = x + (size_t)row * Cc;
    float v[4];
    float s = 0.f, s2 = 0.f;
#pragma unroll
    for (int i = 0; i < 4; ++i) {
        v[i] = xr[threadIdx.x + i * 256];
        s  += v[i];
        s2 += v[i] * v[i];
    }
    s  = block_sum256(s,  sbuf);
    s2 = block_sum256(s2, sbuf);
    const float mu   = s * (1.0f / Cc);
    const float var  = s2 * (1.0f / Cc) - mu * mu;
    const float rstd = rsqrtf(var + 1e-5f);
    _Float16* xo = xn + (size_t)row * Cc;
#pragma unroll
    for (int i = 0; i < 4; ++i) {
        int c = threadIdx.x + i * 256;
        xo[c] = (_Float16)(((v[i] - mu) * rstd) * gamma[c] + beta[c]);
    }
}

// --------------------------- f32 -> f16 convert ----------------------------
__global__ __launch_bounds__(256) void k_cvt(
    const float* __restrict__ src, _Float16* __restrict__ dst, int n)
{
    for (int i = blockIdx.x * 256 + threadIdx.x; i < n; i += gridDim.x * 256)
        dst[i] = (_Float16)src[i];
}

// ------------------------------ QKV GEMM -----------------------------------
// C[m, j] = sum_k xn[m,k] * Wqkv[j,k]   (xn: [4096,1024], Wqkv: [3072,1024])
// Block tile 128(M) x 128(N), K-step 32, double-buffered async-LDS staging.
// 8 waves: wave w -> M-subtile w, iterating 8 N-subtiles. Last step peeled so
// the steady-state loop is branch-free (keeps accumulators pinned).
__global__ __launch_bounds__(256) void k_qkv_gemm(
    const _Float16* __restrict__ A, const _Float16* __restrict__ W,
    _Float16* __restrict__ qf, _Float16* __restrict__ kf,
    _Float16* __restrict__ vf)
{
    __shared__ _Float16 lA[2][128 * 32];   // 2 x 8 KB
    __shared__ _Float16 lB[2][128 * 32];   // 2 x 8 KB
    const int tid  = threadIdx.x;
    const int wave = tid >> 5, lane = tid & 31;
    const int half16 = lane >> 4, l16 = lane & 15;
    const int m0 = wave * 16;
    const int blockM = blockIdx.x * 128;
    const int blockN = blockIdx.y * 128;
    // per-thread staging coords: 512 b128 per tile, 2 per thread
    const int r0 = tid >> 2, kk0 = (tid & 3) << 3;          // chunk 0
    const int r1 = (tid + 256) >> 2, kk1 = kk0;             // chunk 1

    v8f acc[8] = {};

    auto stage = [&](int k0, int buf) {
        async_ld_b128(&A[(size_t)(blockM + r0) * Cc + k0 + kk0],
                      lds_off(&lA[buf][r0 * 32 + kk0]));
        async_ld_b128(&A[(size_t)(blockM + r1) * Cc + k0 + kk1],
                      lds_off(&lA[buf][r1 * 32 + kk1]));
        async_ld_b128(&W[(size_t)(blockN + r0) * Cc + k0 + kk0],
                      lds_off(&lB[buf][r0 * 32 + kk0]));
        async_ld_b128(&W[(size_t)(blockN + r1) * Cc + k0 + kk1],
                      lds_off(&lB[buf][r1 * 32 + kk1]));
    };
    auto compute = [&](int buf) {
        FragH af;   // A layout: lanes0-15 K{0-7,16-23}; lanes16-31 K{8-15,24-31}
        af.q[0] = *(const uint4*)&lA[buf][(m0 + l16) * 32 + half16 * 8];
        af.q[1] = *(const uint4*)&lA[buf][(m0 + l16) * 32 + 16 + half16 * 8];
#pragma unroll
        for (int nt = 0; nt < 8; ++nt) {
            FragH bf; // B layout: lanes0-15 K{0-7,16-23}; lanes16-31 K{8-15,24-31}
            bf.q[0] = *(const uint4*)&lB[buf][(nt * 16 + l16) * 32 + half16 * 8];
            bf.q[1] = *(const uint4*)&lB[buf][(nt * 16 + l16) * 32 + 16 + half16 * 8];
            acc[nt] = wmma_f16(af.v, bf.v, acc[nt]);
        }
    };

    constexpr int T = Cc / 32;
    stage(0, 0);
    for (int t = 0; t < T - 1; ++t) {
        const int buf = t & 1;
        wait_async0();
        __syncthreads();
        stage((t + 1) * 32, buf ^ 1);
        compute(buf);
    }
    wait_async0();
    __syncthreads();
    compute((T - 1) & 1);

    // epilogue: which of q/k/v is block-uniform (blockN is 128-aligned)
    const int which = blockN >> 10;                 // 0=q, 1=k, 2=v
    _Float16* __restrict__ dst = (which == 0) ? qf : ((which == 1) ? kf : vf);
    const float sc = (which == 0) ? 0.125f : 1.0f;  // DH^-0.5 on q
    const int nbase = blockN & (Cc - 1);
#pragma unroll
    for (int nt = 0; nt < 8; ++nt) {
#pragma unroll
        for (int i = 0; i < 8; ++i) {
            int mg  = blockM + m0 + i + half16 * 8;     // [0,4096)
            int rem = nbase + nt * 16 + l16;            // [0,1024)
            int b = mg >> 11, n = mg & (Nn - 1);
            int h = rem >> 6, dh = rem & (DHh - 1);
            size_t o = (((size_t)b * Hh + h) * Nn + n) * DHh + dh;
            dst[o] = (_Float16)(acc[nt][i] * sc);
        }
    }
}

// --------------------------- Flash attention -------------------------------
// grid.y = b*H + h, grid.x covers 128 query rows / block (8 waves x 16 rows).
// Online softmax over key tiles of 32; K staged via async-LDS.
__global__ __launch_bounds__(256) void k_attention(
    const _Float16* __restrict__ qf, const _Float16* __restrict__ kf,
    const _Float16* __restrict__ vf, const float* __restrict__ maskp,
    _Float16* __restrict__ ao)
{
    __shared__ _Float16 lK[32 * 64];        // 4 KB  [key][dh]
    __shared__ _Float16 lVt[64 * 32];       // 4 KB  [dh][key]
    __shared__ _Float16 lP[8 * 16 * 32];    // 8 KB  per-wave P staging
    __shared__ float maskk[32];

    const int tid  = threadIdx.x;
    const int wave = tid >> 5, lane = tid & 31;
    const int half16 = lane >> 4, l16 = lane & 15;
    const int bh = blockIdx.y;              // b*H + h
    const int b  = bh >> 4;
    const int h  = bh & (Hh - 1);
    const int qbase = blockIdx.x * 128 + wave * 16;

    // preload Q fragments (16 rows x 64 dh as two K=32 A-frags)
    FragH afQ[2];
    {
        size_t qrow = ((size_t)bh * Nn + qbase + l16) * DHh;
#pragma unroll
        for (int c = 0; c < 2; ++c) {
            afQ[c].q[0] = *(const uint4*)&qf[qrow + c * 32 + half16 * 8];
            afQ[c].q[1] = *(const uint4*)&qf[qrow + c * 32 + 16 + half16 * 8];
        }
    }
    float mq8[8];
#pragma unroll
    for (int i = 0; i < 8; ++i)
        mq8[i] = maskp[b * Nn + qbase + i + 8 * half16];

    float mrow[8], lsum[8];
    v8f accO[4] = {};
#pragma unroll
    for (int i = 0; i < 8; ++i) { mrow[i] = -1e30f; lsum[i] = 0.f; }

    const int rK = tid >> 3, kkK = (tid & 7) << 3;
    for (int kt = 0; kt < Nn; kt += 32) {
        // stage K tile [32][64] via async global->LDS
        async_ld_b128(&kf[((size_t)bh * Nn + kt + rK) * DHh + kkK],
                      lds_off(&lK[rK * 64 + kkK]));
        // stage V tile transposed -> [dh][key]
        {
            int key = tid >> 3, dh0 = (tid & 7) << 3;
            union { uint4 u; _Float16 hx[8]; } t;
            t.u = *(const uint4*)&vf[((size_t)bh * Nn + kt + key) * DHh + dh0];
#pragma unroll
            for (int e = 0; e < 8; ++e) lVt[(dh0 + e) * 32 + key] = t.hx[e];
        }
        if (tid < 32) maskk[tid] = maskp[b * Nn + kt + tid];
        // prefetch next tiles into cache (global_prefetch_b8)
        if (kt + 32 < Nn) {
            __builtin_prefetch(&kf[((size_t)bh * Nn + kt + 32 + rK) * DHh], 0, 3);
            __builtin_prefetch(&vf[((size_t)bh * Nn + kt + 32 + rK) * DHh], 0, 3);
        }
        wait_async0();
        __syncthreads();

        // S = q @ K^T  (16 x 32 as two 16x16 tiles, K-dim 64 = 2 chained wmma)
        float sv[2][8];
#pragma unroll
        for (int s = 0; s < 2; ++s) {
            v8f sacc = {};
#pragma unroll
            for (int c = 0; c < 2; ++c) {
                FragH bf;
                bf.q[0] = *(const uint4*)&lK[(s * 16 + l16) * 64 + c * 32 + half16 * 8];
                bf.q[1] = *(const uint4*)&lK[(s * 16 + l16) * 64 + c * 32 + 16 + half16 * 8];
                sacc = wmma_f16(afQ[c].v, bf.v, sacc);
            }
            float mk = maskk[s * 16 + l16];
#pragma unroll
            for (int i = 0; i < 8; ++i) {
                float m2 = mq8[i] * mk;
                float am = (m2 == 0.f) ? -111.0f : ((m2 == 1.f) ? 0.f : m2);
                sv[s][i] = sacc[i] + am;
            }
        }

        // online softmax per query row; write P to per-wave LDS (C->A relayout)
#pragma unroll
        for (int i = 0; i < 8; ++i) {
            float mt = fmaxf(sv[0][i], sv[1][i]);
#pragma unroll
            for (int m = 8; m; m >>= 1) mt = fmaxf(mt, __shfl_xor(mt, m, 32));
            float newm  = fmaxf(mrow[i], mt);
            float scale = __expf(mrow[i] - newm);
            float p0 = __expf(sv[0][i] - newm);
            float p1 = __expf(sv[1][i] - newm);
            float ps = p0 + p1;
#pragma unroll
            for (int m = 8; m; m >>= 1) ps += __shfl_xor(ps, m, 32);
            lsum[i] = lsum[i] * scale + ps;
            mrow[i] = newm;
#pragma unroll
            for (int d = 0; d < 4; ++d) accO[d][i] *= scale;
            int M = i + 8 * half16;
            lP[wave * 512 + M * 32 + l16]      = (_Float16)p0;
            lP[wave * 512 + M * 32 + 16 + l16] = (_Float16)p1;
        }

        // O += P(16x32) @ V(32x64)
        FragH ap;
        ap.q[0] = *(const uint4*)&lP[wave * 512 + l16 * 32 + half16 * 8];
        ap.q[1] = *(const uint4*)&lP[wave * 512 + l16 * 32 + 16 + half16 * 8];
#pragma unroll
        for (int d = 0; d < 4; ++d) {
            FragH bv;
            bv.q[0] = *(const uint4*)&lVt[(d * 16 + l16) * 32 + half16 * 8];
            bv.q[1] = *(const uint4*)&lVt[(d * 16 + l16) * 32 + 16 + half16 * 8];
            accO[d] = wmma_f16(ap.v, bv.v, accO[d]);
        }
        __syncthreads();
    }

    // normalize and write attention output [B, N, H*DH] (f16)
#pragma unroll
    for (int i = 0; i < 8; ++i) {
        float rl = 1.0f / lsum[i];
        int query = qbase + i + 8 * half16;
        size_t rowo = ((size_t)b * Nn + query) * Cc + h * DHh;
#pragma unroll
        for (int d = 0; d < 4; ++d)
            ao[rowo + d * 16 + l16] = (_Float16)(accO[d][i] * rl);
    }
}

// ---------------------------- Projection GEMM ------------------------------
// out[m, j] = sum_k ao[m,k] * Wproj[j,k]  -> f32 output
__global__ __launch_bounds__(256) void k_proj_gemm(
    const _Float16* __restrict__ A, const _Float16* __restrict__ W,
    float* __restrict__ out)
{
    __shared__ _Float16 lA[2][128 * 32];
    __shared__ _Float16 lB[2][128 * 32];
    const int tid  = threadIdx.x;
    const int wave = tid >> 5, lane = tid & 31;
    const int half16 = lane >> 4, l16 = lane & 15;
    const int m0 = wave * 16;
    const int blockM = blockIdx.x * 128;
    const int blockN = blockIdx.y * 128;
    const int r0 = tid >> 2, kk0 = (tid & 3) << 3;
    const int r1 = (tid + 256) >> 2, kk1 = kk0;

    v8f acc[8] = {};

    auto stage = [&](int k0, int buf) {
        async_ld_b128(&A[(size_t)(blockM + r0) * Cc + k0 + kk0],
                      lds_off(&lA[buf][r0 * 32 + kk0]));
        async_ld_b128(&A[(size_t)(blockM + r1) * Cc + k0 + kk1],
                      lds_off(&lA[buf][r1 * 32 + kk1]));
        async_ld_b128(&W[(size_t)(blockN + r0) * Cc + k0 + kk0],
                      lds_off(&lB[buf][r0 * 32 + kk0]));
        async_ld_b128(&W[(size_t)(blockN + r1) * Cc + k0 + kk1],
                      lds_off(&lB[buf][r1 * 32 + kk1]));
    };
    auto compute = [&](int buf) {
        FragH af;
        af.q[0] = *(const uint4*)&lA[buf][(m0 + l16) * 32 + half16 * 8];
        af.q[1] = *(const uint4*)&lA[buf][(m0 + l16) * 32 + 16 + half16 * 8];
#pragma unroll
        for (int nt = 0; nt < 8; ++nt) {
            FragH bf;
            bf.q[0] = *(const uint4*)&lB[buf][(nt * 16 + l16) * 32 + half16 * 8];
            bf.q[1] = *(const uint4*)&lB[buf][(nt * 16 + l16) * 32 + 16 + half16 * 8];
            acc[nt] = wmma_f16(af.v, bf.v, acc[nt]);
        }
    };

    constexpr int T = Cc / 32;
    stage(0, 0);
    for (int t = 0; t < T - 1; ++t) {
        const int buf = t & 1;
        wait_async0();
        __syncthreads();
        stage((t + 1) * 32, buf ^ 1);
        compute(buf);
    }
    wait_async0();
    __syncthreads();
    compute((T - 1) & 1);

#pragma unroll
    for (int nt = 0; nt < 8; ++nt) {
#pragma unroll
        for (int i = 0; i < 8; ++i) {
            int mg = blockM + m0 + i + half16 * 8;
            int jg = blockN + nt * 16 + l16;
            out[(size_t)mg * Cc + jg] = acc[nt][i];
        }
    }
}

// ------------------------------- Launcher ----------------------------------
extern "C" void kernel_launch(void* const* d_in, const int* in_sizes, int n_in,
                              void* d_out, int out_size, void* d_ws, size_t ws_size,
                              hipStream_t stream) {
    const float* x      = (const float*)d_in[0];
    const float* mask   = (const float*)d_in[1];
    const float* gamma  = (const float*)d_in[2];
    const float* beta   = (const float*)d_in[3];
    const float* W_qkv  = (const float*)d_in[4];
    const float* W_proj = (const float*)d_in[5];
    float* out = (float*)d_out;

    // workspace layout (f16), total ~48 MB
    _Float16* ws = (_Float16*)d_ws;
    _Float16* xn = ws;                                    // 4096*1024
    _Float16* wq = xn + (size_t)BNr * Cc;                 // 3072*1024
    _Float16* wp = wq + (size_t)C3 * Cc;                  // 1024*1024
    _Float16* qf = wp + (size_t)Cc * Cc;                  // B*H*N*DH
    _Float16* kf = qf + (size_t)Bz * Hh * Nn * DHh;
    _Float16* vf = kf + (size_t)Bz * Hh * Nn * DHh;
    _Float16* ao = vf + (size_t)Bz * Hh * Nn * DHh;       // 4096*1024

    // 1) LayerNorm + f16 cast
    k_layernorm<<<dim3(BNr), dim3(256), 0, stream>>>(x, gamma, beta, xn);

    // 2) convert weights
    k_cvt<<<dim3(2048), dim3(256), 0, stream>>>(W_qkv,  wq, C3 * Cc);
    k_cvt<<<dim3(1024), dim3(256), 0, stream>>>(W_proj, wp, Cc * Cc);

    // 3) QKV projection + head split (+ q scaling)
    k_qkv_gemm<<<dim3(BNr / 128, C3 / 128), dim3(256), 0, stream>>>(
        xn, wq, qf, kf, vf);

    // 4) flash attention per (b,h)
    k_attention<<<dim3(Nn / 128, Bz * Hh), dim3(256), 0, stream>>>(
        qf, kf, vf, mask, ao);

    // 5) output projection -> f32
    k_proj_gemm<<<dim3(BNr / 128, Cc / 128), dim3(256), 0, stream>>>(
        ao, wp, out);
}
